// Quantizer_54142357733732
// MI455X (gfx1250) — matrix operand order repaired and checked by reading, not compiled, not run on hardware
//
#include <hip/hip_runtime.h>

typedef __attribute__((ext_vector_type(16))) _Float16 v16h;
typedef __attribute__((ext_vector_type(8)))  float    v8f;

#define D        64          // embedding dim
#define K_CODES  512         // number of codes
#define TILE_N   16          // codes per WMMA tile
#define N_TILES  (K_CODES / TILE_N)            // 32
#define CHUNK_T  16                            // tiles staged in LDS at once
#define BP_U4_PER_TILE 128                     // 2 kb * 32 lanes * 2 uint4
#define ROWS_PER_WAVE 32                       // two 16-row A tiles per wave

// ---------------------------------------------------------------------------
// Prep 1: pack embed [64][512] f32 -> f16 in WMMA-B operand layout.
// Layout: Bp[T][kb][lane][16 halves], T=code tile (0..31), kb=K block (0..1).
// B fragment (32x16, f16): lane holds column n=lane%16; halves e=0..15 are
// K = kb*32 + (lane<16 ? 0 : 16) + e   (contiguous k order, same as the A fragments).
// ---------------------------------------------------------------------------
__global__ __launch_bounds__(256) void vq_pack_embed(
    const float* __restrict__ embed, _Float16* __restrict__ Bp)
{
    int id = blockIdx.x * blockDim.x + threadIdx.x;   // 0 .. 2047
    if (id >= N_TILES * 2 * 32) return;
    int lane = id & 31;
    int kb   = (id >> 5) & 1;
    int T    = id >> 6;
    int n    = lane & 15;
    int code = T * TILE_N + n;
    int kstart = kb * 32 + ((lane < 16) ? 0 : 16);
    _Float16* dst = Bp + id * 16;
#pragma unroll
    for (int e = 0; e < 16; ++e)
        dst[e] = (_Float16)embed[(kstart + e) * K_CODES + code];
}

// Prep 2: ||e_k||^2 per code (exact f32)
__global__ __launch_bounds__(256) void vq_code_norms(
    const float* __restrict__ embed, float* __restrict__ ee)
{
    int k = blockIdx.x * blockDim.x + threadIdx.x;
    if (k >= K_CODES) return;
    float s = 0.f;
#pragma unroll 8
    for (int d = 0; d < D; ++d) {
        float v = embed[d * K_CODES + k];
        s = fmaf(v, v, s);
    }
    ee[k] = s;
}

// Prep 3: embedT[k][d] = embed[d][k]  (coalesced gather source for q)
__global__ __launch_bounds__(256) void vq_transpose(
    const float* __restrict__ embed, float* __restrict__ embedT)
{
    int id = blockIdx.x * blockDim.x + threadIdx.x;   // 0 .. 32767
    if (id >= D * K_CODES) return;
    int d = id >> 9;           // /512
    int k = id & (K_CODES - 1);
    embedT[k * D + d] = embed[id];
}

// Prep 4: zero the diff accumulator (harness poisons d_out)
__global__ void vq_init_diff(float* diff_out) { diff_out[0] = 0.f; }

// ---------------------------------------------------------------------------
// Main kernel: 8 waves/block, 32 rows per wave (two A tiles share every B
// fragment -> 4 WMMAs per 4 ds_load_b128), full 512-code argmin via WMMA.
// ---------------------------------------------------------------------------
__global__ __launch_bounds__(256) void vq_main(
    const float* __restrict__ x,
    const uint4* __restrict__ Bp,       // packed f16 B operands
    const float* __restrict__ ee,       // ||e_k||^2
    const float* __restrict__ embedT,   // [512][64] f32
    float* __restrict__ q_out,
    float* __restrict__ diff_out,
    float* __restrict__ id_out,
    float  inv_total)
{
    __shared__ uint4 sB[CHUNK_T * BP_U4_PER_TILE];   // 32 KB
    __shared__ float sEE[K_CODES];                   // 2 KB
    __shared__ int   sIdx[8][ROWS_PER_WAVE];         // 1 KB

    const int tid  = threadIdx.x;
    const int wave = tid >> 5;
    const int lane = tid & 31;
    const int half = lane >> 4;       // 0/1 lane half
    const int m    = lane & 15;       // row within A tile / column within B
    const int off  = half * 8;
    const long rowBase = ((long)blockIdx.x * 8 + wave) * ROWS_PER_WAVE;

    for (int i = tid; i < K_CODES; i += 256) sEE[i] = ee[i];

    // ---- build A fragments (16x32 f16, contiguous k order) for both row groups
    //      + exact f32 row sum-of-squares
    v16h a0, a1, a2, a3;
    float ss0 = 0.f, ss1 = 0.f;
    const float* xr0 = x + (rowBase + m) * D;
    const float* xr1 = x + (rowBase + 16 + m) * D;
#pragma unroll
    for (int j = 0; j < 8; ++j) {
        int k0 = 2 * j + 2 * off;   // K pairs (contiguous: k = 16*half + e)
        float u0 = xr0[k0],      u1 = xr0[k0 + 1];
        float v0 = xr0[k0 + 32], v1 = xr0[k0 + 33];
        ss0 += u0 * u0 + u1 * u1 + v0 * v0 + v1 * v1;
        a0[2 * j] = (_Float16)u0;  a0[2 * j + 1] = (_Float16)u1;
        a1[2 * j] = (_Float16)v0;  a1[2 * j + 1] = (_Float16)v1;
        float w0 = xr1[k0],      w1 = xr1[k0 + 1];
        float y0 = xr1[k0 + 32], y1 = xr1[k0 + 33];
        ss1 += w0 * w0 + w1 * w1 + y0 * y0 + y1 * y1;
        a2[2 * j] = (_Float16)w0;  a2[2 * j + 1] = (_Float16)w1;
        a3[2 * j] = (_Float16)y0;  a3[2 * j + 1] = (_Float16)y1;
    }
    float xx0 = ss0 + __shfl_xor(ss0, 16, 32);   // ||x_row||^2, both halves
    float xx1 = ss1 + __shfl_xor(ss1, 16, 32);

    // per-lane base terms: C-matrix VGPR r holds row M = r + 8*half
    float basem0[8], basem1[8];
#pragma unroll
    for (int r = 0; r < 8; ++r) {
        basem0[r] = __shfl(xx0, r + off, 32);
        basem1[r] = __shfl(xx1, r + off, 32);
    }

    float bestv0[8], bestv1[8];
    int   besti0[8], besti1[8];
#pragma unroll
    for (int r = 0; r < 8; ++r) {
        bestv0[r] = 3.4e38f; besti0[r] = 0;
        bestv1[r] = 3.4e38f; besti1[r] = 0;
    }

    for (int chunk = 0; chunk < N_TILES / CHUNK_T; ++chunk) {
        __syncthreads();
        for (int i = tid; i < CHUNK_T * BP_U4_PER_TILE; i += 256)
            sB[i] = Bp[chunk * (CHUNK_T * BP_U4_PER_TILE) + i];
        __syncthreads();

#pragma unroll 2
        for (int t = 0; t < CHUNK_T; ++t) {
            const v16h b0 = *(const v16h*)&sB[(t * 2 + 0) * 64 + lane * 2];
            const v16h b1 = *(const v16h*)&sB[(t * 2 + 1) * 64 + lane * 2];
            v8f c0 = {}, c1 = {};
            c0 = __builtin_amdgcn_wmma_f32_16x16x32_f16(false, a0, false, b0,
                                                        (short)0, c0, false, false);
            c1 = __builtin_amdgcn_wmma_f32_16x16x32_f16(false, a2, false, b0,
                                                        (short)0, c1, false, false);
            c0 = __builtin_amdgcn_wmma_f32_16x16x32_f16(false, a1, false, b1,
                                                        (short)0, c0, false, false);
            c1 = __builtin_amdgcn_wmma_f32_16x16x32_f16(false, a3, false, b1,
                                                        (short)0, c1, false, false);
            const int code = (chunk * CHUNK_T + t) * TILE_N + m;
            const float en = sEE[code];
#pragma unroll
            for (int r = 0; r < 8; ++r) {
                float d0 = fmaf(-2.f, c0[r], basem0[r] + en);
                bool b0w = d0 < bestv0[r];     // ascending code -> lowest idx on tie
                besti0[r] = b0w ? code : besti0[r];
                bestv0[r] = b0w ? d0   : bestv0[r];
                float d1 = fmaf(-2.f, c1[r], basem1[r] + en);
                bool b1w = d1 < bestv1[r];
                besti1[r] = b1w ? code : besti1[r];
                bestv1[r] = b1w ? d1   : bestv1[r];
            }
        }
    }

    // ---- butterfly all-reduce over the 16 lanes holding each row half
#pragma unroll
    for (int s = 1; s < 16; s <<= 1) {
#pragma unroll
        for (int r = 0; r < 8; ++r) {
            float ov = __shfl_xor(bestv0[r], s, 32);
            int   oi = __shfl_xor(besti0[r], s, 32);
            if (ov < bestv0[r] || (ov == bestv0[r] && oi < besti0[r])) {
                bestv0[r] = ov; besti0[r] = oi;
            }
            float ow = __shfl_xor(bestv1[r], s, 32);
            int   oj = __shfl_xor(besti1[r], s, 32);
            if (ow < bestv1[r] || (ow == bestv1[r] && oj < besti1[r])) {
                bestv1[r] = ow; besti1[r] = oj;
            }
        }
    }
    if (m == 0) {
#pragma unroll
        for (int r = 0; r < 8; ++r) {
            sIdx[wave][r + off]      = besti0[r];
            sIdx[wave][16 + r + off] = besti1[r];
        }
    }
    __syncthreads();

    // ---- emd_id (float-valued indices in the flat f32 output buffer)
    id_out[rowBase + lane] = (float)sIdx[wave][lane];

    // ---- q gather (coalesced float4 from embedT) + commitment-loss partial
    const int qi = sIdx[wave][lane];             // one row per lane
    const float4* et = (const float4*)(embedT + (long)qi * D);
    const float4* xv = (const float4*)(x + (rowBase + lane) * D);
    float4* qv = (float4*)(q_out + (rowBase + lane) * D);
    float ds = 0.f;
#pragma unroll
    for (int j = 0; j < 16; ++j) {
        float4 e4 = et[j];
        float4 x4 = xv[j];
        qv[j] = e4;
        float d0 = e4.x - x4.x, d1 = e4.y - x4.y;
        float d2 = e4.z - x4.z, d3 = e4.w - x4.w;
        ds += d0 * d0 + d1 * d1 + d2 * d2 + d3 * d3;
    }
#pragma unroll
    for (int s = 1; s < 32; s <<= 1) ds += __shfl_xor(ds, s, 32);
    if (lane == 0) atomicAdd(diff_out, ds * inv_total);
}

// ---------------------------------------------------------------------------
extern "C" void kernel_launch(void* const* d_in, const int* in_sizes, int n_in,
                              void* d_out, int out_size, void* d_ws, size_t ws_size,
                              hipStream_t stream)
{
    const float* x     = (const float*)d_in[0];   // [N, 64] flat
    const float* embed = (const float*)d_in[1];   // [64, 512]
    const int N = in_sizes[0] / D;                // 131072 rows

    float* out      = (float*)d_out;
    float* q_out    = out;                        // N*64 floats
    float* diff_out = out + (long)N * D;          // 1 float
    float* id_out   = diff_out + 1;               // N floats (index values)

    // workspace layout
    char* ws = (char*)d_ws;
    _Float16* Bp_h   = (_Float16*)ws;                       // 64 KB packed B
    float*    ee     = (float*)(ws + 65536);                // 2 KB norms
    float*    embedT = (float*)(ws + 65536 + 2048);         // 128 KB transpose

    vq_pack_embed<<<8,   256, 0, stream>>>(embed, Bp_h);
    vq_code_norms<<<2,   256, 0, stream>>>(embed, ee);
    vq_transpose <<<128, 256, 0, stream>>>(embed, embedT);
    vq_init_diff <<<1,   1,   0, stream>>>(diff_out);

    const float inv_total = 1.0f / ((float)N * (float)D);
    vq_main<<<N / (8 * ROWS_PER_WAVE), 256, 0, stream>>>(
        x, (const uint4*)Bp_h, ee, embedT, q_out, diff_out, id_out, inv_total);
}
